// DeBERTaDisentangledAttention_84000970375731
// MI455X (gfx1250) — matrix operand order repaired and checked by reading, not compiled, not run on hardware
//
#include <hip/hip_runtime.h>
#include <hip/hip_bf16.h>

typedef __attribute__((ext_vector_type(16))) _Float16 v16h;
typedef __attribute__((ext_vector_type(8)))  float    v8f;

namespace cfg {
constexpr int  N  = 8, S = 384, E = 768, H = 12, D = 64, P = 384, TP = 768;
constexpr long NSE  = (long)N * S * E;       // 2,359,296
constexpr long EE   = (long)E * E;           // 589,824
constexpr long ATT  = (long)N * H * S * S;   // 14,155,776
constexpr long B2SZ = (long)H * S * TP;      // 3,538,944
constexpr float SCALE = 0.03608439182435161f; // 1/sqrt(768)
}

// ---------------------------------------------------------------- converts
__global__ void cvt_f32_f16(const float* __restrict__ src,
                            _Float16* __restrict__ dst, long n) {
  long i = (long)blockIdx.x * blockDim.x + threadIdx.x;
  if (i < n) dst[i] = (_Float16)src[i];
}

// ---------------------------------------------------------------- fragments
// A 16x32 f16 layout: lane owns row M=lane&15; VGPR j -> K=16*hf+2j{,+1} (same K order as B).
// Contiguous 32B at k0+16*hf.
__device__ inline v16h load_frag_a(const _Float16* __restrict__ Ab, int k0, int hf) {
  v16h a;
#pragma unroll
  for (int j = 0; j < 8; ++j) {
    const int ka = 16 * hf + 2 * j;
    const _Float16* pa = Ab + k0 + ka;
    a[2 * j]     = pa[0];
    a[2 * j + 1] = pa[1];
  }
  return a;
}

// B 32x16 f16 layout: lane owns col N=lane&15; lanes 0-15 hold K=0..15, 16-31 hold 16..31.
template <bool BK1>
__device__ inline v16h load_frag_b(const _Float16* __restrict__ Bb, int k0, int hf,
                                   int ldbk) {
  v16h b;
#pragma unroll
  for (int j = 0; j < 8; ++j) {
    const int kb = 16 * hf + 2 * j;
    const _Float16* pb = Bb + (long)(k0 + kb) * (BK1 ? 1 : ldbk);
    b[2 * j]     = pb[0];
    b[2 * j + 1] = BK1 ? pb[1] : pb[ldbk];
  }
  return b;
}

#define WMMA_F16(A_, B_, C_)                                                   \
  __builtin_amdgcn_wmma_f32_16x16x32_f16(false, (A_), false, (B_), (short)0,   \
                                         (C_), false, false)

// ---------------------------------------------------------------- WMMA GEMM
// C[i,j] = sum_k A[i*lda + k] * B[j*ldbj + k*ldbk]  (+ bias[j])
// One wave computes a 32x32 super-tile: 2 A-frags x 2 B-frags -> 4 WMMA per K step,
// with next-step fragments loaded ahead of the WMMAs (double-buffered registers).
template <bool CF16, bool HASBIAS, bool BK1>
__global__ void gemm_wmma(const _Float16* __restrict__ A,
                          const _Float16* __restrict__ B,
                          const float* __restrict__ bias,
                          float* __restrict__ Cf,
                          _Float16* __restrict__ Ch,
                          int M, int Ncol, int K,
                          int lda, int ldbj, int ldbk, int ldc,
                          int batchH,
                          long sAn, long sAh, long sBn, long sBh,
                          long sCn, long sCh) {
  const int wave = threadIdx.x >> 5;
  const int lane = threadIdx.x & 31;
  const int r  = lane & 15;
  const int hf = lane >> 4;

  const int superN = Ncol >> 5;
  const int superM = M >> 5;
  const int st = blockIdx.x * (blockDim.x >> 5) + wave;
  if (st >= superM * superN) return;        // wave-uniform
  const int stm = st / superN;
  const int stn = st % superN;

  const int b = blockIdx.y;
  const int n = b / batchH;
  const int h = b % batchH;

  const _Float16* Ab0 = A + n * sAn + h * sAh + (long)(stm * 32 + r) * lda;
  const _Float16* Ab1 = Ab0 + (long)16 * lda;
  const _Float16* Bb0 = B + n * sBn + h * sBh + (long)(stn * 32 + r) * ldbj;
  const _Float16* Bb1 = Bb0 + (long)16 * ldbj;

  v8f acc00 = {}, acc01 = {}, acc10 = {}, acc11 = {};

  // prologue: fragments for k0 = 0
  v16h a0 = load_frag_a(Ab0, 0, hf);
  v16h a1 = load_frag_a(Ab1, 0, hf);
  v16h b0 = load_frag_b<BK1>(Bb0, 0, hf, ldbk);
  v16h b1 = load_frag_b<BK1>(Bb1, 0, hf, ldbk);

  for (int k0 = 0;;) {
    const int kn = k0 + 32;
    // current-step fragments
    const v16h ca0 = a0, ca1 = a1, cb0 = b0, cb1 = b1;
    // issue next-step loads before the WMMAs so they overlap the matrix pipe
    if (kn < K) {
      __builtin_prefetch(Ab0 + kn + 32, 0, 3);
      __builtin_prefetch(Bb0 + (long)(kn + 32) * (BK1 ? 1 : ldbk), 0, 3);
      a0 = load_frag_a(Ab0, kn, hf);
      a1 = load_frag_a(Ab1, kn, hf);
      b0 = load_frag_b<BK1>(Bb0, kn, hf, ldbk);
      b1 = load_frag_b<BK1>(Bb1, kn, hf, ldbk);
    }
    // 4 back-to-back WMMAs (32x32 super-tile)
    acc00 = WMMA_F16(ca0, cb0, acc00);
    acc01 = WMMA_F16(ca0, cb1, acc01);
    acc10 = WMMA_F16(ca1, cb0, acc10);
    acc11 = WMMA_F16(ca1, cb1, acc11);
    k0 = kn;
    if (k0 >= K) break;
  }

  // epilogue: C layout VGPR i -> M = i + 8*hf, N = lane&15
  const long Cbase = n * sCn + h * sCh;
  const int col0 = stn * 32 + r;
  const int col1 = col0 + 16;
  const float bv0 = HASBIAS ? bias[col0] : 0.0f;
  const float bv1 = HASBIAS ? bias[col1] : 0.0f;
#pragma unroll
  for (int i = 0; i < 8; ++i) {
    const int m0 = stm * 32 + i + 8 * hf;
    const int m1 = m0 + 16;
    const float v00 = acc00[i] + bv0, v01 = acc01[i] + bv1;
    const float v10 = acc10[i] + bv0, v11 = acc11[i] + bv1;
    if (CF16) {
      Ch[Cbase + (long)m0 * ldc + col0] = (_Float16)v00;
      Ch[Cbase + (long)m0 * ldc + col1] = (_Float16)v01;
      Ch[Cbase + (long)m1 * ldc + col0] = (_Float16)v10;
      Ch[Cbase + (long)m1 * ldc + col1] = (_Float16)v11;
    } else {
      Cf[Cbase + (long)m0 * ldc + col0] = v00;
      Cf[Cbase + (long)m0 * ldc + col1] = v01;
      Cf[Cbase + (long)m1 * ldc + col0] = v10;
      Cf[Cbase + (long)m1 * ldc + col1] = v11;
    }
  }
}

// ---------------------------------------------------------------- pos gather
// scores[n][h,q,k] += B2[h,q, k - q + P]   (B2 holds qh . kp_flat^T for one n)
__global__ void add_pos_scores(float* __restrict__ sc,          // + n*H*S*S
                               const float* __restrict__ B2) {  // [H,S,2P]
  using namespace cfg;
  const long idx = (long)blockIdx.x * blockDim.x + threadIdx.x; // over H*S*S
  const int k = (int)(idx % S);
  const long t = idx / S;
  const int q = (int)(t % S);
  const int h = (int)(t / S);
  const int r = k - q + P;                                      // in [1, 2P)
  sc[idx] += B2[((long)h * S + q) * TP + r];
}

// ---------------------------------------------------------------- softmax
// One wave per row of S=384 (12 elems/lane). Writes f32 attn (d_out) + f16 copy.
__global__ void softmax_rows(float* __restrict__ scores,
                             _Float16* __restrict__ attn16,
                             const int* __restrict__ mask) {
  using namespace cfg;
  const int wave = threadIdx.x >> 5;
  const int lane = threadIdx.x & 31;
  const long row = (long)blockIdx.x * 8 + wave;   // row in [0, N*H*S)
  const int n = (int)(row / ((long)H * S));
  float* p = scores + row * S;
  const int* mrow = mask + (long)n * S;

  float vals[12];
  float mx = -INFINITY;
#pragma unroll
  for (int i = 0; i < 12; ++i) {
    const int k = lane + 32 * i;
    float s = p[k] * SCALE;
    if (mrow[k] == 0) s = -INFINITY;
    vals[i] = s;
    mx = fmaxf(mx, s);
  }
#pragma unroll
  for (int off = 16; off > 0; off >>= 1) mx = fmaxf(mx, __shfl_xor(mx, off, 32));

  float sum = 0.0f;
#pragma unroll
  for (int i = 0; i < 12; ++i) {
    vals[i] = __expf(vals[i] - mx);
    sum += vals[i];
  }
#pragma unroll
  for (int off = 16; off > 0; off >>= 1) sum += __shfl_xor(sum, off, 32);
  const float inv = 1.0f / sum;

#pragma unroll
  for (int i = 0; i < 12; ++i) {
    const int k = lane + 32 * i;
    const float a = vals[i] * inv;
    p[k] = a;
    attn16[row * S + k] = (_Float16)a;
  }
}

// ---------------------------------------------------------------- launcher
extern "C" void kernel_launch(void* const* d_in, const int* in_sizes, int n_in,
                              void* d_out, int out_size, void* d_ws, size_t ws_size,
                              hipStream_t stream) {
  using namespace cfg;
  const float* value = (const float*)d_in[0];
  const float* key   = (const float*)d_in[1];
  const float* query = (const float*)d_in[2];
  const int*   mask  = (const int*)d_in[3];
  const float* Wq  = (const float*)d_in[4];  const float* bq  = (const float*)d_in[5];
  const float* Wkc = (const float*)d_in[6];  const float* bkc = (const float*)d_in[7];
  const float* Wkp = (const float*)d_in[8];  const float* bkp = (const float*)d_in[9];
  const float* Wv  = (const float*)d_in[10]; const float* bv  = (const float*)d_in[11];
  const float* Wfc = (const float*)d_in[12]; const float* bfc = (const float*)d_in[13];
  const float* pos = (const float*)d_in[14];

  char* w = (char*)d_ws;
  size_t off = 0;
  auto take = [&](size_t bytes) -> void* {
    off = (off + 255) & ~(size_t)255;
    void* p = w + off;
    off += bytes;
    return p;
  };
  _Float16* q16    = (_Float16*)take(NSE * 2);
  _Float16* k16    = (_Float16*)take(NSE * 2);
  _Float16* v16    = (_Float16*)take(NSE * 2);
  _Float16* Wq16   = (_Float16*)take(EE * 2);
  _Float16* Wkc16  = (_Float16*)take(EE * 2);
  _Float16* Wkp16  = (_Float16*)take(EE * 2);
  _Float16* Wv16   = (_Float16*)take(EE * 2);
  _Float16* Wfc16  = (_Float16*)take(EE * 2);
  _Float16* pe16   = (_Float16*)take(EE * 2);
  _Float16* qh16   = (_Float16*)take(NSE * 2);
  _Float16* kc16   = (_Float16*)take(NSE * 2);
  _Float16* vp16   = (_Float16*)take(NSE * 2);
  _Float16* kpf16  = (_Float16*)take(EE * 2);
  _Float16* outh16 = (_Float16*)take(NSE * 2);
  _Float16* attn16 = (_Float16*)take(ATT * 2);
  float*    B2     = (float*)   take(B2SZ * 4);

  float* out_f  = (float*)d_out;           // [N,S,E]
  float* attn_f = out_f + NSE;             // [N,H,S,S]

  // --- 1. f32 -> f16 conversions
  auto cvt = [&](const float* s, _Float16* d, long n) {
    cvt_f32_f16<<<dim3((unsigned)((n + 255) / 256)), dim3(256), 0, stream>>>(s, d, n);
  };
  cvt(query, q16, NSE);  cvt(key, k16, NSE);  cvt(value, v16, NSE);
  cvt(Wq, Wq16, EE);  cvt(Wkc, Wkc16, EE);  cvt(Wkp, Wkp16, EE);
  cvt(Wv, Wv16, EE);  cvt(Wfc, Wfc16, EE);  cvt(pos, pe16, EE);

  const long SE = (long)S * E, SS = (long)S * S;

  // --- 2. projections: [N*S,E] x [E,E]^T + bias -> f16
  // super-tiles = 96*24 = 2304 -> 288 blocks of 8 waves
  gemm_wmma<true, true, true><<<dim3(288, 1), dim3(256), 0, stream>>>(
      q16, Wq16, bq, nullptr, qh16, N * S, E, E, E, E, 1, E, 1, 0, 0, 0, 0, 0, 0);
  gemm_wmma<true, true, true><<<dim3(288, 1), dim3(256), 0, stream>>>(
      k16, Wkc16, bkc, nullptr, kc16, N * S, E, E, E, E, 1, E, 1, 0, 0, 0, 0, 0, 0);
  gemm_wmma<true, true, true><<<dim3(288, 1), dim3(256), 0, stream>>>(
      v16, Wv16, bv, nullptr, vp16, N * S, E, E, E, E, 1, E, 1, 0, 0, 0, 0, 0, 0);
  // kp_flat = pos_emb @ Wkp^T + bkp : [2P,E]; super-tiles = 24*24 = 576 -> 72 blocks
  gemm_wmma<true, true, true><<<dim3(72, 1), dim3(256), 0, stream>>>(
      pe16, Wkp16, bkp, nullptr, kpf16, TP, E, E, E, E, 1, E, 1, 0, 0, 0, 0, 0, 0);

  // --- 3. content scores: batched (n,h): [S,D] x [S,D]^T -> f32 into d_out attn region
  // super-tiles per batch = 12*12 = 144 -> 18 blocks; grid.y = 96
  gemm_wmma<false, false, true><<<dim3(18, N * H), dim3(256), 0, stream>>>(
      qh16, kc16, nullptr, attn_f, nullptr, S, S, D, E, E, 1, S, H,
      SE, (long)D, SE, (long)D, (long)H * SS, SS);

  // --- 4. position scores per n (reuses one B2 buffer): B2 = qh[n] . kp_flat^T, then gather-add
  for (int n = 0; n < N; ++n) {
    // super-tiles per h = 12*24 = 288 -> 36 blocks; grid.y = 12
    gemm_wmma<false, false, true><<<dim3(36, H), dim3(256), 0, stream>>>(
        qh16 + (long)n * SE, kpf16, nullptr, B2, nullptr, S, TP, D, E, E, 1, TP, H,
        0, (long)D, 0, (long)D, 0, (long)S * TP);
    add_pos_scores<<<dim3((unsigned)(H * SS / 256)), dim3(256), 0, stream>>>(
        attn_f + (long)n * H * SS, B2);
  }

  // --- 5. softmax: N*H*S rows / 8 rows per block = 4608 blocks
  softmax_rows<<<dim3(4608), dim3(256), 0, stream>>>(attn_f, attn16, mask);

  // --- 6. attn @ V: batched (n,h): [S,S] x [S,D] -> f16; B indexed (j=d, k) with ldbk=E
  // super-tiles per batch = 12*2 = 24 -> 3 blocks; grid.y = 96
  gemm_wmma<true, false, false><<<dim3(3, N * H), dim3(256), 0, stream>>>(
      attn16, vp16, nullptr, nullptr, outh16, S, D, S, S, 1, E, E, H,
      (long)H * SS, SS, SE, (long)D, SE, (long)D);

  // --- 7. final FC: [N*S,E] x [E,E]^T + bias -> f32 out
  gemm_wmma<false, true, true><<<dim3(288, 1), dim3(256), 0, stream>>>(
      outh16, Wfc16, bfc, out_f, nullptr, N * S, E, E, E, E, 1, E, 1, 0, 0, 0, 0, 0, 0);
}
